// S5Classifier_24352464569726
// MI455X (gfx1250) — matrix operand order repaired and checked by reading, not compiled, not run on hardware
//
#include <hip/hip_runtime.h>

// ---------------------------------------------------------------------------
// S5 classifier forward for gfx1250 (MI455X).
// GEMMs: v_wmma_f32_16x16x32_bf16, A fragments from bf16 LDS via ds_load_b128,
// B fragments from pre-swizzled bf16 weights in global (L2-resident) via
// global_load_b128. All elementwise stages (LN/GELU/residual) fused into the
// WMMA tile epilogues. Length-8192 complex recurrences are wave-parallel
// (lane-chunked affine-map composition + shfl_up scan).
// ---------------------------------------------------------------------------

#define H_    128
#define P_    64
#define L_    8192
#define B_    16
#define CIN_  64
#define NTOK_ 131072            // B_*L_

#define ACT_LD 132              // f32 LDS stride for [32][128] tiles
#define XALD   264              // bf16 LDS stride for [32][256] tiles
#define FY_LD  136              // bf16 LDS stride for [32][128] tiles
#define XIN_LD 72               // bf16 LDS stride for [32][64] input tiles

typedef __attribute__((ext_vector_type(16))) __bf16 v16bf;
typedef __attribute__((ext_vector_type(8)))  __bf16 v8bf;
typedef __attribute__((ext_vector_type(8)))  float  v8f;

__device__ __forceinline__ float gelu_f(float x) {
    return 0.5f * x * (1.0f + erff(x * 0.70710678118654752f));
}

// Swizzled weight layout: 16 contiguous bf16 per (k-group, n):
//   idx(k,n) = ((k/16)*N + n)*16 + k%16
__device__ __forceinline__ long swidx(int k, int n, int N) {
    return ((long)(k >> 4) * N + n) * 16 + (k & 15);
}

// ---- WMMA fragment helpers (wave32; A/B use contiguous k = 16*half + s) ----------
// A: 16x32 bf16 from bf16 LDS, row-major, lda multiple of 8.
// Lane (m, half): elements k0+kh..+15 (kh = 16*half) -> two 16B ds loads.
__device__ __forceinline__ v16bf load_frag_a_bf(const __bf16* a, int lda, int k0) {
    const int lane = threadIdx.x & 31;
    const int m    = lane & 15;
    const int kh   = (lane & 16) ? 16 : 0;
    const __bf16* p = a + (long)m * lda + k0 + kh;
    v8bf lo = *(const v8bf*)p;
    v8bf hi = *(const v8bf*)(p + 8);
    return __builtin_shufflevector(lo, hi, 0, 1, 2, 3, 4, 5, 6, 7,
                                   8, 9, 10, 11, 12, 13, 14, 15);
}

// B: 32x16 bf16 from swizzled global weights: lane's 16 values contiguous.
__device__ __forceinline__ v16bf load_frag_b_sw(const __bf16* bsw, int N, int n0, int k0) {
    const int lane = threadIdx.x & 31;
    const int n    = n0 + (lane & 15);
    const long grp = (long)(k0 >> 4) + ((lane & 16) ? 1 : 0);
    const __bf16* p = bsw + (grp * N + n) * 16;
    v8bf lo = *(const v8bf*)p;
    v8bf hi = *(const v8bf*)(p + 8);
    return __builtin_shufflevector(lo, hi, 0, 1, 2, 3, 4, 5, 6, 7,
                                   8, 9, 10, 11, 12, 13, 14, 15);
}

// C/D: VGPR v -> (M = mb+v, N = lane&15)
__device__ __forceinline__ void store_frag_T(float* dst, int ldd, const v8f& c) {
    const int lane = threadIdx.x & 31;
    const int n    = lane & 15;
    const int mb   = (lane & 16) ? 8 : 0;
#pragma unroll
    for (int v = 0; v < 8; ++v) dst[(long)n * ldd + mb + v] = c[v];
}

__device__ __forceinline__ void store_frag_bf16(__bf16* dst, int ldd, const v8f& c) {
    const int lane = threadIdx.x & 31;
    const int n    = lane & 15;
    const int mb   = (lane & 16) ? 8 : 0;
#pragma unroll
    for (int v = 0; v < 8; ++v) dst[(long)(mb + v) * ldd + n] = (__bf16)c[v];
}

#define WMMA_BF16(acc, af, bf) \
    acc = __builtin_amdgcn_wmma_f32_16x16x32_bf16(false, af, false, bf, (short)0, acc, false, false)

// Fused LayerNorm over 128 columns; 8 lanes per row, 16 cols per lane.
__device__ __forceinline__ void ln128(const float* src, int sld, float* dst, long dld,
                                      const float* g, const float* bb) {
    const int row = threadIdx.x >> 3;
    const int c0  = (threadIdx.x & 7) * 16;
    float s = 0.f, s2 = 0.f;
#pragma unroll
    for (int j = 0; j < 16; ++j) { float v = src[row * sld + c0 + j]; s += v; s2 += v * v; }
#pragma unroll
    for (int off = 1; off < 8; off <<= 1) { s += __shfl_xor(s, off); s2 += __shfl_xor(s2, off); }
    const float mean = s * (1.0f / 128.0f);
    const float var  = s2 * (1.0f / 128.0f) - mean * mean;
    const float rstd = rsqrtf(var + 1e-5f);
#pragma unroll
    for (int j = 0; j < 16; ++j) {
        int c = c0 + j;
        float v = src[row * sld + c];
        dst[(long)row * dld + c] = (v - mean) * rstd * g[c] + bb[c];
    }
}

// LayerNorm emitting both f32 (for residual) and bf16 (as next GEMM's A).
__device__ __forceinline__ void ln128_dual(const float* src, int sld, float* dstf, int dfd,
                                           __bf16* dstb, int dbd,
                                           const float* g, const float* bb) {
    const int row = threadIdx.x >> 3;
    const int c0  = (threadIdx.x & 7) * 16;
    float s = 0.f, s2 = 0.f;
#pragma unroll
    for (int j = 0; j < 16; ++j) { float v = src[row * sld + c0 + j]; s += v; s2 += v * v; }
#pragma unroll
    for (int off = 1; off < 8; off <<= 1) { s += __shfl_xor(s, off); s2 += __shfl_xor(s2, off); }
    const float mean = s * (1.0f / 128.0f);
    const float var  = s2 * (1.0f / 128.0f) - mean * mean;
    const float rstd = rsqrtf(var + 1e-5f);
#pragma unroll
    for (int j = 0; j < 16; ++j) {
        int c = c0 + j;
        float v = src[row * sld + c];
        float o = (v - mean) * rstd * g[c] + bb[c];
        dstf[row * dfd + c] = o;
        dstb[row * dbd + c] = (__bf16)o;
    }
}

// ---------------------------------------------------------------------------
// Once: swizzle W_in -> bf16 fragment layout
// ---------------------------------------------------------------------------
__global__ __launch_bounds__(256) void wsetup_kernel(const float* __restrict__ win,
                                                     __bf16* __restrict__ winsw) {
    for (int i = threadIdx.x; i < CIN_ * H_; i += 256) {
        int k = i >> 7, n = i & 127;
        winsw[swidx(k, n, H_)] = (__bf16)win[(long)k * H_ + n];
    }
}

// ---------------------------------------------------------------------------
// Per-layer: discretize (Lbar, BbarT) + build swizzled bf16 weights
// ---------------------------------------------------------------------------
__global__ __launch_bounds__(256) void setup_kernel(const float* __restrict__ lam_re,
                                                    const float* __restrict__ lam_im,
                                                    const float* __restrict__ Bre,
                                                    const float* __restrict__ Bim,
                                                    const float* __restrict__ Cre,
                                                    const float* __restrict__ Cim,
                                                    const float* __restrict__ lstep,
                                                    const float* __restrict__ wenc,
                                                    const float* __restrict__ wdec,
                                                    float* __restrict__ lbr, float* __restrict__ lbi,
                                                    __bf16* __restrict__ bbtswr,
                                                    __bf16* __restrict__ bbtswi,
                                                    __bf16* __restrict__ cmatsw,
                                                    __bf16* __restrict__ wencsw,
                                                    __bf16* __restrict__ wdecsw) {
    __shared__ float s_cr[P_], s_ci[P_];
    const int tid = threadIdx.x;
    if (tid < P_) {
        float step = expf(lstep[tid]);
        float lr = lam_re[tid], li = lam_im[tid];
        float er = expf(lr * step);
        float Lr = er * cosf(li * step), Li = er * sinf(li * step);
        lbr[tid] = Lr; lbi[tid] = Li;
        float den = lr * lr + li * li;
        s_cr[tid] = ((Lr - 1.f) * lr + Li * li) / den;
        s_ci[tid] = (Li * lr - (Lr - 1.f) * li) / den;
    }
    __syncthreads();
    // BbarT[h][p] = coef(p)*Bc[p][h], bf16 swizzled (K=H_, N=P_)
    for (int i = tid; i < P_ * H_; i += 256) {
        int p = i >> 7, hh = i & 127;
        float br = Bre[p * H_ + hh], bi = Bim[p * H_ + hh];
        long idx = swidx(hh, p, P_);
        bbtswr[idx] = (__bf16)(s_cr[p] * br - s_ci[p] * bi);
        bbtswi[idx] = (__bf16)(s_cr[p] * bi + s_ci[p] * br);
    }
    // Fused real C matrix: out_re = [xf_re|xf_im|xb_re|xb_im] @ cmat  (K=256,N=H_)
    for (int i = tid; i < 256 * H_; i += 256) {
        int k = i >> 7, hh = i & 127;
        int seg = k >> 6, pp = k & 63;
        float v;
        if      (seg == 0) v =  Cre[hh * 128 + pp];
        else if (seg == 1) v = -Cim[hh * 128 + pp];
        else if (seg == 2) v =  Cre[hh * 128 + 64 + pp];
        else               v = -Cim[hh * 128 + 64 + pp];
        cmatsw[swidx(k, hh, H_)] = (__bf16)v;
    }
    for (int i = tid; i < H_ * 2 * H_; i += 256) {
        int k = i >> 8, n = i & 255;
        wencsw[swidx(k, n, 2 * H_)] = (__bf16)wenc[(long)k * 2 * H_ + n];
    }
    for (int i = tid; i < H_ * H_; i += 256) {
        int k = i >> 7, n = i & 127;
        wdecsw[swidx(k, n, H_)] = (__bf16)wdec[(long)k * H_ + n];
    }
}

// ---------------------------------------------------------------------------
// K1: h = x^T @ W_in + b_in      x:[B,CIN,L] -> h:[B*L,H]
// ---------------------------------------------------------------------------
__global__ __launch_bounds__(256) void inproj_kernel(const float* __restrict__ x,
                                                     const __bf16* __restrict__ winsw,
                                                     const float* __restrict__ bin,
                                                     float* __restrict__ hbuf) {
    __shared__ __align__(16) __bf16 abf[32 * XIN_LD];
    const int blk = blockIdx.x;
    const int b   = blk >> 8;              // L_/32 = 256 blocks per batch
    const int l0  = (blk & 255) * 32;
    const float* xb = x + (size_t)b * CIN_ * L_ + l0;
    // stage x tile transposed: abf[l][c] = x[b][c][l0+l]  (coalesced along l)
    for (int i = threadIdx.x; i < CIN_ * 32; i += 256) {
        int c = i >> 5, l = i & 31;
        abf[l * XIN_LD + c] = (__bf16)xb[(size_t)c * L_ + l];
    }
    __syncthreads();
    const int w    = threadIdx.x >> 5;
    const int lane = threadIdx.x & 31;
    const int ln_  = lane & 15;
    const int mb   = (lane & 16) ? 8 : 0;
#pragma unroll
    for (int j = 0; j < 2; ++j) {
        const int tile = w + 8 * j;        // 16 tiles: 2 (M) x 8 (N)
        const int mt = tile >> 3, nt = tile & 7;
        v8f acc = {};
#pragma unroll
        for (int k = 0; k < CIN_; k += 32) {
            v16bf af = load_frag_a_bf(&abf[(mt * 16) * XIN_LD], XIN_LD, k);
            v16bf bf = load_frag_b_sw(winsw, H_, nt * 16, k);
            WMMA_BF16(acc, af, bf);
        }
        const int col = nt * 16 + ln_;
        const float bv = bin[col];
#pragma unroll
        for (int v = 0; v < 8; ++v) {
            int l = l0 + mt * 16 + mb + v;
            hbuf[((size_t)b * L_ + l) * H_ + col] = acc[v] + bv;
        }
    }
}

// ---------------------------------------------------------------------------
// K2: fx = LN(h); Bu = fx @ BbarT   (Bu written [b*P+p][t] for the scan)
// ---------------------------------------------------------------------------
__global__ __launch_bounds__(256) void pre_kernel(const float* __restrict__ hbuf,
                                                  float* __restrict__ fxg,
                                                  float* __restrict__ bure,
                                                  float* __restrict__ buim,
                                                  const __bf16* __restrict__ bbtswr,
                                                  const __bf16* __restrict__ bbtswi,
                                                  const float* __restrict__ g,
                                                  const float* __restrict__ bta) {
    __shared__ float act[32 * ACT_LD];
    __shared__ __align__(16) __bf16 fxbf[32 * FY_LD];
    __shared__ float bs_re[P_ * 33];
    __shared__ float bs_im[P_ * 33];
    const int r0 = blockIdx.x * 32;
    const int b  = r0 >> 13;
    const int t0 = r0 & (L_ - 1);
    const size_t gbase = (size_t)r0 * H_;
    __builtin_prefetch(hbuf + gbase, 0, 3);
    for (int i = threadIdx.x; i < 32 * H_; i += 256)
        act[(i >> 7) * ACT_LD + (i & 127)] = hbuf[gbase + i];
    __syncthreads();
    ln128(act, ACT_LD, act, ACT_LD, g, bta);          // in-place LN -> fx
    __syncthreads();
    for (int i = threadIdx.x; i < 32 * H_; i += 256) {
        float v = act[(i >> 7) * ACT_LD + (i & 127)];
        fxg[gbase + i] = v;                           // persist fx for post pass
        fxbf[(i >> 7) * FY_LD + (i & 127)] = (__bf16)v;
    }
    __syncthreads();
    const int w = threadIdx.x >> 5;
#pragma unroll
    for (int j = 0; j < 2; ++j) {
        const int tile = w + 8 * j;                   // 16 tiles: {re,im} x 2M x 4N
        const int sel  = tile >> 3;
        const int t2   = tile & 7;
        const int mt = t2 >> 2, nt = t2 & 3;
        const __bf16* Bb = sel ? bbtswi : bbtswr;
        v8f acc = {};
#pragma unroll
        for (int k = 0; k < H_; k += 32) {
            v16bf af = load_frag_a_bf(&fxbf[(mt * 16) * FY_LD], FY_LD, k);
            v16bf bf = load_frag_b_sw(Bb, P_, nt * 16, k);
            WMMA_BF16(acc, af, bf);
        }
        float* dst = (sel ? bs_im : bs_re) + (nt * 16) * 33 + mt * 16;
        store_frag_T(dst, 33, acc);                   // LDS as [p][t]
    }
    __syncthreads();
    for (int i = threadIdx.x; i < P_ * 32; i += 256) { // coalesced along t
        int p = i >> 5, tt = i & 31;
        size_t go = ((size_t)(b * P_ + p)) * L_ + t0 + tt;
        bure[go] = bs_re[p * 33 + tt];
        buim[go] = bs_im[p * 33 + tt];
    }
}

// ---------------------------------------------------------------------------
// K3: wave-parallel complex recurrence x[t] = Lbar*x[t-1] + Bu[t].
// One wave per (b,p); lane i owns a 256-element chunk; 5-step shfl_up
// composition scan; dir=1 runs reversed and may execute in place.
// ---------------------------------------------------------------------------
__global__ __launch_bounds__(256) void scan_kernel(const float* __restrict__ sre,
                                                   const float* __restrict__ sim,
                                                   float* __restrict__ dre,
                                                   float* __restrict__ dim_,
                                                   const float* __restrict__ lbr,
                                                   const float* __restrict__ lbi,
                                                   int dir) {
    const int wid  = blockIdx.x * 8 + (threadIdx.x >> 5);   // 0..B*P-1
    const int lane = threadIdx.x & 31;
    const int p    = wid & (P_ - 1);
    const float ar = lbr[p], ai = lbi[p];
    const size_t ro = (size_t)wid * L_;
    const int c0 = lane * 256;
    float Ar = ar, Ai = ai;                  // chunk transition Lbar^256
#pragma unroll
    for (int q = 0; q < 8; ++q) { float nr = Ar * Ar - Ai * Ai; Ai = 2.f * Ar * Ai; Ar = nr; }
    float Br = 0.f, Bi = 0.f;                // pass 1: chunk offset from zero state
    for (int j = 0; j < 256; ++j) {
        int t = dir ? (L_ - 1 - (c0 + j)) : (c0 + j);
        float u = sre[ro + t], vv = sim[ro + t];
        float nb = Br * ar - Bi * ai + u;
        Bi = Br * ai + Bi * ar + vv;
        Br = nb;
    }
#pragma unroll
    for (int off = 1; off < 32; off <<= 1) { // inclusive lane scan of (A,B)
        float pAr = __shfl_up(Ar, off);
        float pAi = __shfl_up(Ai, off);
        float pBr = __shfl_up(Br, off);
        float pBi = __shfl_up(Bi, off);
        if (lane >= off) {
            float nbr = pBr * Ar - pBi * Ai + Br;
            float nbi = pBr * Ai + pBi * Ar + Bi;
            Br = nbr; Bi = nbi;
            float nar = pAr * Ar - pAi * Ai;
            float nai = pAr * Ai + pAi * Ar;
            Ar = nar; Ai = nai;
        }
    }
    float srr = __shfl_up(Br, 1), sii = __shfl_up(Bi, 1);   // exclusive prefix
    if (lane == 0) { srr = 0.f; sii = 0.f; }
    float xr = srr, xi = sii;                // pass 2: replay, emit states
    for (int j = 0; j < 256; ++j) {
        int t = dir ? (L_ - 1 - (c0 + j)) : (c0 + j);
        float u = sre[ro + t], vv = sim[ro + t];
        float nr = xr * ar - xi * ai + u;
        xi = xr * ai + xi * ar + vv;
        xr = nr;
        dre[ro + t] = xr;
        dim_[ro + t] = xi;
    }
}

// ---------------------------------------------------------------------------
// K4: s5 = xs @ Cmat; a = gelu(s5 + D*fx) + fx; fy = LN(a);
//     e = fy@W_enc; t = v*gelu(g); blk = t@W_dec + fy; h = LN(blk + res)
// ---------------------------------------------------------------------------
__global__ __launch_bounds__(256) void post_kernel(float* __restrict__ hbuf,
                                                   const float* __restrict__ fxg,
                                                   const float* __restrict__ xfre,
                                                   const float* __restrict__ xfim,
                                                   const float* __restrict__ xbre,
                                                   const float* __restrict__ xbim,
                                                   const __bf16* __restrict__ cmatsw,
                                                   const float* __restrict__ Dl,
                                                   const float* __restrict__ ffg,
                                                   const float* __restrict__ ffb,
                                                   const __bf16* __restrict__ wencsw,
                                                   const __bf16* __restrict__ wdecsw,
                                                   const float* __restrict__ outg,
                                                   const float* __restrict__ outb) {
    __shared__ __align__(16) __bf16 abf[32 * XALD];   // xs, then e (bf16 A staging)
    __shared__ __align__(16) __bf16 fybf[32 * FY_LD]; // fy bf16, then t bf16
    __shared__ float act[32 * ACT_LD];                // fx -> a -> res -> blk sum
    __shared__ float act2[32 * ACT_LD];               // fy f32
    const int r0 = blockIdx.x * 32;
    const int b  = r0 >> 13;
    const int t0 = r0 & (L_ - 1);
    const size_t gbase = (size_t)r0 * H_;
    __builtin_prefetch(fxg + gbase, 0, 3);
    __builtin_prefetch(hbuf + gbase, 0, 3);
    // stage xs = [xf_re|xf_im|xb_re|xb_im] -> abf[32][256], coalesced along t
    for (int i = threadIdx.x; i < 4 * P_ * 32; i += 256) {
        int seg = i >> 11, rem = i & 2047;
        int p = rem >> 5, r = rem & 31;
        const float* src = (seg == 0) ? xfre : ((seg == 1) ? xfim : ((seg == 2) ? xbre : xbim));
        abf[r * XALD + (seg << 6) + p] = (__bf16)src[((size_t)(b * P_ + p)) * L_ + t0 + r];
    }
    // stage fx tile (coalesced) -> act
    for (int i = threadIdx.x; i < 32 * H_; i += 256)
        act[(i >> 7) * ACT_LD + (i & 127)] = fxg[gbase + i];
    __syncthreads();
    const int w    = threadIdx.x >> 5;
    const int lane = threadIdx.x & 31;
    const int ln_  = lane & 15;
    const int mb   = (lane & 16) ? 8 : 0;
    // --- s5: [32,256]x[256,128]; epilogue: + D*fx, gelu, + fx (in place) ---
#pragma unroll
    for (int j = 0; j < 2; ++j) {
        const int tile = w + 8 * j;
        const int mt = tile >> 3, nt = tile & 7;
        v8f acc = {};
#pragma unroll
        for (int k = 0; k < 256; k += 32) {
            v16bf af = load_frag_a_bf(&abf[(mt * 16) * XALD], XALD, k);
            v16bf bf = load_frag_b_sw(cmatsw, H_, nt * 16, k);
            WMMA_BF16(acc, af, bf);
        }
        const int col = nt * 16 + ln_;
        const float dv = Dl[col];
#pragma unroll
        for (int v = 0; v < 8; ++v) {
            int lr = mt * 16 + mb + v;
            float fxv = act[lr * ACT_LD + col];            // fx (owner element)
            float val = acc[v] + dv * fxv;
            act[lr * ACT_LD + col] = gelu_f(val) + fxv;    // a
        }
    }
    __syncthreads();
    ln128_dual(act, ACT_LD, act2, ACT_LD, fybf, FY_LD, ffg, ffb);  // fy (f32 + bf16)
    __syncthreads();
    // --- e = fy @ W_enc : [32,128]x[128,256] -> abf (bf16) ---
#pragma unroll
    for (int j = 0; j < 4; ++j) {
        const int tile = w + 8 * j;
        const int mt = tile >> 4, nt = tile & 15;
        v8f acc = {};
#pragma unroll
        for (int k = 0; k < H_; k += 32) {
            v16bf af = load_frag_a_bf(&fybf[(mt * 16) * FY_LD], FY_LD, k);
            v16bf bf = load_frag_b_sw(wencsw, 2 * H_, nt * 16, k);
            WMMA_BF16(acc, af, bf);
        }
        store_frag_bf16(&abf[(mt * 16) * XALD + nt * 16], XALD, acc);
    }
    __syncthreads();
    // --- GEGLU: t = v * gelu(g) -> fybf; stage residual h -> act ---
    for (int i = threadIdx.x; i < 32 * H_; i += 256) {
        int r = i >> 7, c = i & 127;
        float vv = (float)abf[r * XALD + c];
        float gg = (float)abf[r * XALD + 128 + c];
        fybf[r * FY_LD + c] = (__bf16)(vv * gelu_f(gg));
        act[r * ACT_LD + c] = hbuf[gbase + i];             // res (coalesced)
    }
    __syncthreads();
    // --- blk = t @ W_dec + fy + res (in-place owner update of act) ---
#pragma unroll
    for (int j = 0; j < 2; ++j) {
        const int tile = w + 8 * j;
        const int mt = tile >> 3, nt = tile & 7;
        v8f acc = {};
#pragma unroll
        for (int k = 0; k < H_; k += 32) {
            v16bf af = load_frag_a_bf(&fybf[(mt * 16) * FY_LD], FY_LD, k);
            v16bf bf = load_frag_b_sw(wdecsw, H_, nt * 16, k);
            WMMA_BF16(acc, af, bf);
        }
        const int col = nt * 16 + ln_;
#pragma unroll
        for (int v = 0; v < 8; ++v) {
            int lr = mt * 16 + mb + v;
            float val = acc[v] + act2[lr * ACT_LD + col] + act[lr * ACT_LD + col];
            act[lr * ACT_LD + col] = val;
        }
    }
    __syncthreads();
    ln128(act, ACT_LD, hbuf + gbase, H_, outg, outb);      // h for next layer
}

// ---------------------------------------------------------------------------
// K5/K6: mean pool over L, then 2-layer MLP head
// ---------------------------------------------------------------------------
__global__ __launch_bounds__(128) void pool_kernel(const float* __restrict__ hbuf,
                                                   float* __restrict__ pooled) {
    const int b = blockIdx.x, hh = threadIdx.x;
    float s = 0.f;
    const size_t base = (size_t)b * L_ * H_ + hh;
    for (int l = 0; l < L_; ++l) s += hbuf[base + (size_t)l * H_];
    pooled[b * H_ + hh] = s * (1.0f / (float)L_);
}

__global__ __launch_bounds__(256) void head_kernel(const float* __restrict__ pooled,
                                                   const float* __restrict__ Wc1,
                                                   const float* __restrict__ bc1,
                                                   const float* __restrict__ Wc2,
                                                   const float* __restrict__ bc2,
                                                   float* __restrict__ out) {
    __shared__ float z[B_ * 64];
    for (int i = threadIdx.x; i < B_ * 64; i += 256) {
        int bb = i >> 6, j = i & 63;
        float s = bc1[j];
        for (int hh = 0; hh < H_; ++hh) s += pooled[bb * H_ + hh] * Wc1[hh * 64 + j];
        z[i] = fmaxf(s, 0.f);
    }
    __syncthreads();
    for (int i = threadIdx.x; i < B_ * 3; i += 256) {
        int bb = i / 3, n = i % 3;
        float s = bc2[n];
        for (int j = 0; j < 64; ++j) s += z[bb * 64 + j] * Wc2[j * 3 + n];
        out[i] = s;
    }
}

// ---------------------------------------------------------------------------
extern "C" void kernel_launch(void* const* d_in, const int* in_sizes, int n_in,
                              void* d_out, int out_size, void* d_ws, size_t ws_size,
                              hipStream_t stream) {
    (void)in_sizes; (void)n_in; (void)out_size; (void)ws_size;
    const float* x        = (const float*)d_in[0];
    const float* W_in     = (const float*)d_in[1];
    const float* b_in     = (const float*)d_in[2];
    const float* attn_g   = (const float*)d_in[3];
    const float* attn_b   = (const float*)d_in[4];
    const float* Lam_re   = (const float*)d_in[5];
    const float* Lam_im   = (const float*)d_in[6];
    const float* B_re     = (const float*)d_in[7];
    const float* B_im     = (const float*)d_in[8];
    const float* C_re     = (const float*)d_in[9];
    const float* C_im     = (const float*)d_in[10];
    const float* Dm       = (const float*)d_in[11];
    const float* log_step = (const float*)d_in[12];
    const float* ff_g     = (const float*)d_in[13];
    const float* ff_b     = (const float*)d_in[14];
    const float* W_enc    = (const float*)d_in[15];
    const float* W_dec    = (const float*)d_in[16];
    const float* out_g    = (const float*)d_in[17];
    const float* out_b    = (const float*)d_in[18];
    const float* Wc1      = (const float*)d_in[19];
    const float* bc1      = (const float*)d_in[20];
    const float* Wc2      = (const float*)d_in[21];
    const float* bc2      = (const float*)d_in[22];

    // Workspace layout (f32 region then bf16 region). Total ~269 MB.
    float* wsf    = (float*)d_ws;
    float* hbuf   = wsf;                                   // [NTOK,H]
    float* fxb    = hbuf + (size_t)NTOK_ * H_;             // [NTOK,H]
    float* bure   = fxb  + (size_t)NTOK_ * H_;             // [B*P,L] (becomes xb_re)
    float* buim   = bure + (size_t)NTOK_ * P_;             // [B*P,L] (becomes xb_im)
    float* xfre   = buim + (size_t)NTOK_ * P_;             // [B*P,L]
    float* xfim   = xfre + (size_t)NTOK_ * P_;             // [B*P,L]
    float* lbr    = xfim + (size_t)NTOK_ * P_;             // [P]
    float* lbi    = lbr + P_;
    float* pooled = lbi + P_;                              // [B,H]
    // bf16 swizzled weights (32B-aligned: offsets are multiples of 8 floats)
    __bf16* winsw   = (__bf16*)(pooled + B_ * H_);
    __bf16* bbtswr  = winsw  + CIN_ * H_;                  // 8192
    __bf16* bbtswi  = bbtswr + H_ * P_;                    // 8192
    __bf16* cmatsw  = bbtswi + H_ * P_;                    // 32768
    __bf16* wencsw  = cmatsw + 256 * H_;                   // 32768
    __bf16* wdecsw  = wencsw + H_ * 2 * H_;                // 16384

    wsetup_kernel<<<1, 256, 0, stream>>>(W_in, winsw);
    inproj_kernel<<<NTOK_ / 32, 256, 0, stream>>>(x, winsw, b_in, hbuf);

    for (int i = 0; i < 2; ++i) {
        setup_kernel<<<1, 256, 0, stream>>>(Lam_re + i * P_, Lam_im + i * P_,
                                            B_re + (size_t)i * P_ * H_, B_im + (size_t)i * P_ * H_,
                                            C_re + (size_t)i * H_ * 2 * P_, C_im + (size_t)i * H_ * 2 * P_,
                                            log_step + i * P_,
                                            W_enc + (size_t)i * H_ * 2 * H_,
                                            W_dec + (size_t)i * H_ * H_,
                                            lbr, lbi, bbtswr, bbtswi, cmatsw, wencsw, wdecsw);
        pre_kernel<<<NTOK_ / 32, 256, 0, stream>>>(hbuf, fxb, bure, buim, bbtswr, bbtswi,
                                                   attn_g + i * H_, attn_b + i * H_);
        scan_kernel<<<(B_ * P_) / 8, 256, 0, stream>>>(bure, buim, xfre, xfim, lbr, lbi, 0);
        scan_kernel<<<(B_ * P_) / 8, 256, 0, stream>>>(bure, buim, bure, buim, lbr, lbi, 1); // in place -> xb
        post_kernel<<<NTOK_ / 32, 256, 0, stream>>>(hbuf, fxb, xfre, xfim, bure, buim, cmatsw,
                                                    Dm + i * H_, ff_g + i * H_, ff_b + i * H_,
                                                    wencsw, wdecsw, out_g + i * H_, out_b + i * H_);
    }

    pool_kernel<<<B_, 128, 0, stream>>>(hbuf, pooled);
    head_kernel<<<1, 256, 0, stream>>>(pooled, Wc1, bc1, Wc2, bc2, (float*)d_out);
}
